// Decoder_50775103373400
// MI455X (gfx1250) — matrix operand order repaired and checked by reading, not compiled, not run on hardware
//
#include <hip/hip_runtime.h>
#include <hip/hip_bf16.h>

typedef __attribute__((ext_vector_type(16))) _Float16 v16h;
typedef __attribute__((ext_vector_type(8)))  _Float16 v8h;
typedef __attribute__((ext_vector_type(8)))  float    v8f;

#define DEV __device__ __forceinline__

static constexpr int BATCH    = 16384;
static constexpr int OBS_LEN  = 16;
static constexpr int PRED_LEN = 25;
// weight tile table (each tile = one 16x16x32 WMMA B operand, 1024B in LDS)
static constexpr int T_IH0 = 0;   // 3 ktiles x 8 ntiles
static constexpr int T_HH0 = 24;  // 8
static constexpr int T_IH1 = 32;  // 8
static constexpr int T_HH1 = 40;  // 8
static constexpr int T_MLP = 48;  // 6
static constexpr int T_FC  = 54;  // 2 (K padded 16->32 with zeros)
static constexpr int NWT   = 56;

DEV v16h cat16(v8h lo, v8h hi) {
  v16h r;
#pragma unroll
  for (int i = 0; i < 8; ++i) { r[i] = lo[i]; r[i + 8] = hi[i]; }
  return r;
}

// A fragment (16x32 f16): lane holds row (lane&15); K-halves split by lane>>4.
DEV v16h load_afrag(const _Float16* base, int stride_h, int kt, int lane) {
  const _Float16* p = base + (lane & 15) * stride_h + kt * 32 + ((lane >> 4) * 8);
  return cat16(*(const v8h*)p, *(const v8h*)(p + 16));
}

// B fragment pre-packed in LDS: 32 lanes x 8 dwords contiguous per tile.
DEV v16h load_bfrag(const unsigned* s_w, int tile, int lane) {
  const _Float16* p = (const _Float16*)(s_w + tile * 256 + lane * 8);
  return cat16(*(const v8h*)p, *(const v8h*)(p + 8));
}

DEV v8f wmma_f16(v16h a, v16h b, v8f c) {
  return __builtin_amdgcn_wmma_f32_16x16x32_f16(false, a, false, b, (short)0, c,
                                                false, false);
}

DEV v8f bcast8(float x) {
  v8f r;
#pragma unroll
  for (int i = 0; i < 8; ++i) r[i] = x;
  return r;
}

DEV float sigm(float x) { return 1.f / (1.f + __expf(-x)); }
DEV float tanh_f(float x) {
  x = fminf(8.f, fmaxf(-8.f, x));
  float e = __expf(2.f * x);
  return (e - 1.f) / (e + 1.f);
}

// Wave-private LDS ordering: DS ops from one wave complete in order at the LDS
// unit (CDNA5 ISA §11), so cross-lane store->load within a wave only needs the
// compiler to preserve DS instruction order — no s_barrier, no dscnt drain.
DEV void wave_lds_fence() {
  __builtin_amdgcn_fence(__ATOMIC_RELEASE, "wavefront");
  __builtin_amdgcn_wave_barrier();
  __builtin_amdgcn_fence(__ATOMIC_ACQUIRE, "wavefront");
}

// store a 16x16 f32 C/D tile into LDS rows as f16 (cols 16j..16j+15)
DEV void store_tile(_Float16* base, int stride_h, int j, int lane, v8f t) {
  const int row0 = (lane >> 4) * 8, col = j * 16 + (lane & 15);
#pragma unroll
  for (int r = 0; r < 8; ++r) base[(row0 + r) * stride_h + col] = (_Float16)t[r];
}

__global__ __launch_bounds__(128)
void decoder_lstm_kernel(const float* __restrict__ obs,   const float* __restrict__ latent,
                         const float* __restrict__ W_fc,  const float* __restrict__ b_fc,
                         const float* __restrict__ W_ih0, const float* __restrict__ W_hh0,
                         const float* __restrict__ b_ih0, const float* __restrict__ b_hh0,
                         const float* __restrict__ W_ih1, const float* __restrict__ W_hh1,
                         const float* __restrict__ b_ih1, const float* __restrict__ b_hh1,
                         const float* __restrict__ W_mlp, const float* __restrict__ b_mlp,
                         float* __restrict__ out)
{
  __shared__ __align__(16) unsigned s_w[NWT * 256];        // 57344 B, B-fragment layout f16
  __shared__ float s_b0[128], s_b1[128], s_bmlp[96], s_bfc[32];
  __shared__ __align__(16) _Float16 s_x [4][16 * 96];      // per-wave X rows (f16)
  __shared__ __align__(16) _Float16 s_h0[4][16 * 32];
  __shared__ __align__(16) _Float16 s_h1[4][16 * 32];

  const int tid = threadIdx.x, lane = tid & 31, wave = tid >> 5;

  // ---- pack all weights into LDS as WMMA B fragments (f16) ----
  for (int idx = tid; idx < NWT * 256; idx += 128) {
    const int t = idx >> 8, rem = idx & 255, ln = rem >> 3, v = rem & 7;
    const float* W; int kdim, nt, kt = 0;
    if      (t < 24) { W = W_ih0; kt = t >> 3; nt = t & 7; kdim = 96; }
    else if (t < 32) { W = W_hh0; nt = t - 24; kdim = 32; }
    else if (t < 40) { W = W_ih1; nt = t - 32; kdim = 32; }
    else if (t < 48) { W = W_hh1; nt = t - 40; kdim = 32; }
    else if (t < 54) { W = W_mlp; nt = t - 48; kdim = 32; }
    else             { W = W_fc;  nt = t - 54; kdim = 16; }
    const int k = kt * 32 + ((ln >> 4) << 3) + (v << 1) + ((v >> 2) << 3);   // B element (k,n) = W[n][k]
    const int n = nt * 16 + (ln & 15);
    const float w0 = (k     < kdim) ? W[n * kdim + k]     : 0.f;
    const float w1 = (k + 1 < kdim) ? W[n * kdim + k + 1] : 0.f;
    const unsigned short u0 = __builtin_bit_cast(unsigned short, (_Float16)w0);
    const unsigned short u1 = __builtin_bit_cast(unsigned short, (_Float16)w1);
    s_w[idx] = (unsigned)u0 | ((unsigned)u1 << 16);
  }
  if (tid < 128) { s_b0[tid] = b_ih0[tid] + b_hh0[tid]; s_b1[tid] = b_ih1[tid] + b_hh1[tid]; }
  if (tid < 96)  s_bmlp[tid] = b_mlp[tid];
  if (tid < 32)  s_bfc[tid]  = b_fc[tid];

  // ---- per-wave 16-row batch tile ----
  const int bbase = blockIdx.x * 64 + wave * 16;
  _Float16* xw  = s_x[wave];
  _Float16* h0w = s_h0[wave];
  _Float16* h1w = s_h1[wave];

  // stage X0 = obs_s[-1] and latent (K-padded to 32) into LDS rows
  const float* xsrc = obs + (size_t)(OBS_LEN - 1) * BATCH * 96 + (size_t)bbase * 96;
  for (int i = lane; i < 16 * 96; i += 32) xw[i] = (_Float16)xsrc[i];
  for (int i = lane; i < 16 * 32; i += 32) {
    const int r = i >> 5, c = i & 31;
    h0w[i] = (c < 16) ? (_Float16)latent[(size_t)(bbase + r) * 16 + c] : (_Float16)0.f;
  }
  __syncthreads();   // cross-wave: weight image must be complete

  const int col = lane & 15, row0 = (lane >> 4) * 8;

  // h_init = latent @ W_fc^T + b_fc  (one K-chunk, 2 N-tiles)
  const v16h alat = load_afrag(h0w, 32, 0, lane);
  v8f hin[2];
#pragma unroll
  for (int j = 0; j < 2; ++j)
    hin[j] = wmma_f16(alat, load_bfrag(s_w, T_FC + j, lane), bcast8(s_bfc[j * 16 + col]));
  wave_lds_fence();                              // cross-lane WAR on s_h0 before overwrite
#pragma unroll
  for (int j = 0; j < 2; ++j) { store_tile(h0w, 32, j, lane, hin[j]); store_tile(h1w, 32, j, lane, hin[j]); }
  v8f c0a[2], c1a[2];
#pragma unroll
  for (int j = 0; j < 2; ++j) { c0a[j] = bcast8(0.f); c1a[j] = bcast8(0.f); }
  wave_lds_fence();

#pragma unroll 1
  for (int t = 0; t < PRED_LEN; ++t) {
    const v16h ax0 = load_afrag(xw, 96, 0, lane);
    const v16h ax1 = load_afrag(xw, 96, 1, lane);
    const v16h ax2 = load_afrag(xw, 96, 2, lane);
    const v16h ah0 = load_afrag(h0w, 32, 0, lane);
    const v16h ah1 = load_afrag(h1w, 32, 0, lane);

    // ---- layer 0: gates = X@W_ih0^T + H0@W_hh0^T + b0, process column-halves ----
    auto gates0 = [&](int nt) -> v8f {
      v8f acc = bcast8(s_b0[nt * 16 + col]);
      acc = wmma_f16(ax0, load_bfrag(s_w, T_IH0 +  0 + nt, lane), acc);
      acc = wmma_f16(ax1, load_bfrag(s_w, T_IH0 +  8 + nt, lane), acc);
      acc = wmma_f16(ax2, load_bfrag(s_w, T_IH0 + 16 + nt, lane), acc);
      acc = wmma_f16(ah0, load_bfrag(s_w, T_HH0 +      nt, lane), acc);
      return acc;
    };
#pragma unroll
    for (int j = 0; j < 2; ++j) {
      const v8f gi = gates0(j), gf = gates0(2 + j), gg = gates0(4 + j), go = gates0(6 + j);
      v8f cn, hn;
#pragma unroll
      for (int e = 0; e < 8; ++e) {
        const float cv = sigm(gf[e]) * c0a[j][e] + sigm(gi[e]) * tanh_f(gg[e]);
        cn[e] = cv;
        hn[e] = sigm(go[e]) * tanh_f(cv);
      }
      c0a[j] = cn;
      store_tile(h0w, 32, j, lane, hn);
    }
    wave_lds_fence();

    // ---- layer 1 ----
    const v16h ah0n = load_afrag(h0w, 32, 0, lane);
    auto gates1 = [&](int nt) -> v8f {
      v8f acc = bcast8(s_b1[nt * 16 + col]);
      acc = wmma_f16(ah0n, load_bfrag(s_w, T_IH1 + nt, lane), acc);
      acc = wmma_f16(ah1,  load_bfrag(s_w, T_HH1 + nt, lane), acc);
      return acc;
    };
#pragma unroll
    for (int j = 0; j < 2; ++j) {
      const v8f gi = gates1(j), gf = gates1(2 + j), gg = gates1(4 + j), go = gates1(6 + j);
      v8f cn, hn;
#pragma unroll
      for (int e = 0; e < 8; ++e) {
        const float cv = sigm(gf[e]) * c1a[j][e] + sigm(gi[e]) * tanh_f(gg[e]);
        cn[e] = cv;
        hn[e] = sigm(go[e]) * tanh_f(cv);
      }
      c1a[j] = cn;
      store_tile(h1w, 32, j, lane, hn);
    }
    wave_lds_fence();

    // ---- MLP head: curr = H1 @ W_mlp^T + b_mlp; write out + feed back as next X ----
    const v16h ah1n = load_afrag(h1w, 32, 0, lane);
    float* outp = out + (size_t)t * (BATCH * 96) + (size_t)bbase * 96;
#pragma unroll
    for (int nt = 0; nt < 6; ++nt) {
      v8f acc = bcast8(s_bmlp[nt * 16 + col]);
      acc = wmma_f16(ah1n, load_bfrag(s_w, T_MLP + nt, lane), acc);
      const int c = nt * 16 + col;
#pragma unroll
      for (int r = 0; r < 8; ++r) {
        outp[(size_t)(row0 + r) * 96 + c] = acc[r];
        xw[(row0 + r) * 96 + c] = (_Float16)acc[r];
      }
    }
    wave_lds_fence();
  }
}

extern "C" void kernel_launch(void* const* d_in, const int* in_sizes, int n_in,
                              void* d_out, int out_size, void* d_ws, size_t ws_size,
                              hipStream_t stream) {
  (void)in_sizes; (void)n_in; (void)d_ws; (void)ws_size; (void)out_size;
  const float* obs    = (const float*)d_in[0];
  const float* latent = (const float*)d_in[1];
  // d_in[2] = pred_len scalar (known constant 25)
  const float* W_fc   = (const float*)d_in[3];
  const float* b_fc   = (const float*)d_in[4];
  const float* W_ih0  = (const float*)d_in[5];
  const float* W_hh0  = (const float*)d_in[6];
  const float* b_ih0  = (const float*)d_in[7];
  const float* b_hh0  = (const float*)d_in[8];
  const float* W_ih1  = (const float*)d_in[9];
  const float* W_hh1  = (const float*)d_in[10];
  const float* b_ih1  = (const float*)d_in[11];
  const float* b_hh1  = (const float*)d_in[12];
  const float* W_mlp  = (const float*)d_in[13];
  const float* b_mlp  = (const float*)d_in[14];
  float* out = (float*)d_out;

  dim3 grid(BATCH / 64), block(128);   // 4 waves/block, 16 batch rows per wave
  decoder_lstm_kernel<<<grid, block, 0, stream>>>(obs, latent, W_fc, b_fc,
                                                  W_ih0, W_hh0, b_ih0, b_hh0,
                                                  W_ih1, W_hh1, b_ih1, b_hh1,
                                                  W_mlp, b_mlp, out);
}
